// FastSurf_33775622815975
// MI455X (gfx1250) — matrix operand order repaired and checked by reading, not compiled, not run on hardware
//
#include <hip/hip_runtime.h>

typedef __attribute__((ext_vector_type(16))) _Float16 v16h;
typedef __attribute__((ext_vector_type(8)))  _Float16 v8h;
typedef __attribute__((ext_vector_type(8)))  float    v8f;
typedef __attribute__((ext_vector_type(4)))  float    v4f;

#define RES   160
#define GC    12

#define SA    72      // halves per staging row (input, 64 used + pad)
#define SB    136     // halves per staging row (hidden, 128 used + pad)
#define WAVES 8

// Packed B-fragment arena in d_ws. One fragment = 32 lanes x 16 halves = 512 halves = 1KB.
// Fragment index within a matrix: f = nt*ktiles + kt.
#define FRAG_DW1 0
#define FRAG_DW2 (8 * 512)
#define FRAG_DW3 (40 * 512)
#define FRAG_RW1 (44 * 512)
#define FRAG_RW2 (60 * 512)
#define FRAG_RW3 (92 * 512)
#define TOTAL_HALVES (96 * 512)

// ---------------------------------------------------------------------------
// Pack fp32 weights [K][N] (row-major) into WMMA B-fragment layout (f16),
// zero-padding K up to 32*ktiles and N up to 16*ntiles.
// B layout (16-bit, 32x16 per fragment): lane L holds column n = nt*16 + (L&15);
// half h holds K = kt*32 + (L>=16 ? 8 : 0) + (h<8 ? h : h+8).
// ---------------------------------------------------------------------------
__global__ void pack_weights(const float* __restrict__ dW1, const float* __restrict__ dW2,
                             const float* __restrict__ dW3, const float* __restrict__ rW1,
                             const float* __restrict__ rW2, const float* __restrict__ rW3,
                             _Float16* __restrict__ ws) {
  int t = blockIdx.x * blockDim.x + threadIdx.x;
  if (t >= TOTAL_HALVES) return;
  const float* Wp; int K, Nn, ktiles, base;
  if      (t < FRAG_DW2) { Wp = dW1; K = 12;  Nn = 128; ktiles = 1; base = FRAG_DW1; }
  else if (t < FRAG_DW3) { Wp = dW2; K = 128; Nn = 128; ktiles = 4; base = FRAG_DW2; }
  else if (t < FRAG_RW1) { Wp = dW3; K = 128; Nn = 1;   ktiles = 4; base = FRAG_DW3; }
  else if (t < FRAG_RW2) { Wp = rW1; K = 41;  Nn = 128; ktiles = 2; base = FRAG_RW1; }
  else if (t < FRAG_RW3) { Wp = rW2; K = 128; Nn = 128; ktiles = 4; base = FRAG_RW2; }
  else                   { Wp = rW3; K = 128; Nn = 3;   ktiles = 4; base = FRAG_RW3; }
  int r  = t - base;
  int f  = r >> 9;             // fragment index
  int q  = r & 511;
  int L  = q >> 4;             // lane 0..31
  int h  = q & 15;             // half index within v16h
  int kt = f % ktiles, nt = f / ktiles;
  int v  = h >> 1, lh = h & 1;
  int k  = kt * 32 + ((L >= 16) ? 8 : 0) + (h < 8 ? h : h + 8);
  int n  = nt * 16 + (L & 15);
  float val = (k < K && n < Nn) ? Wp[(long long)k * Nn + n] : 0.0f;
  ws[t] = (_Float16)val;
}

// ---------------------------------------------------------------------------
// Fragment loads / stores
// A layout (16-bit, 16x32): lane L holds row m = L&15;
//   halves 0..7  = K(kt*32 + (L<16?0:8)   .. +7)   -> one b128
//   halves 8..15 = K(kt*32 + (L<16?16:24) .. +7)   -> one b128
// ---------------------------------------------------------------------------
__device__ __forceinline__ v16h load_a_frag(const _Float16* row, int kt, int lane) {
  int koff = kt * 32 + ((lane >= 16) ? 8 : 0);
  v8h lo = *(const v8h*)(row + koff);
  v8h hi = *(const v8h*)(row + koff + 16);
  v16h a;
#pragma unroll
  for (int i = 0; i < 8; i++) { a[i] = lo[i]; a[8 + i] = hi[i]; }
  return a;
}

__device__ __forceinline__ v16h load_b_frag(const _Float16* __restrict__ ws, int off, int lane) {
  return *(const v16h*)(ws + off + lane * 16);
}

__device__ __forceinline__ v8f wmma_f16(v16h a, v16h b, v8f c) {
  return __builtin_amdgcn_wmma_f32_16x16x32_f16(false, a, false, b, (short)0, c, false, false);
}

// D layout (f32 16x16): lane L holds col n = L&15; VGPR v holds row m = v + (L>=16?8:0)
__device__ __forceinline__ void store_act(_Float16* buf, int stride, int nt, int lane,
                                          v8f c, const float* bias) {
  int col = nt * 16 + (lane & 15);
  float b = bias[col];
  int mhi = (lane >= 16) ? 8 : 0;
#pragma unroll
  for (int v = 0; v < 8; v++) {
    float f = c[v] + b;
    f = f > 0.0f ? f : 0.0f;
    buf[(v + mhi) * stride + col] = (_Float16)f;
  }
}

template <int KT>
__device__ __forceinline__ void mlp_layer(const _Float16* bin, int sin_,
                                          _Float16* bout, int sout,
                                          const _Float16* __restrict__ ws, int fragbase,
                                          const float* bias, int lane) {
  const _Float16* row = bin + (lane & 15) * sin_;
  v16h A[KT];
#pragma unroll
  for (int kt = 0; kt < KT; kt++) A[kt] = load_a_frag(row, kt, lane);
  __builtin_amdgcn_wave_barrier();
#pragma unroll
  for (int nt = 0; nt < 8; nt++) {
    v8f c;
#pragma unroll
    for (int i = 0; i < 8; i++) c[i] = 0.0f;
#pragma unroll
    for (int kt = 0; kt < KT; kt++) {
      v16h b = load_b_frag(ws, fragbase + (nt * KT + kt) * 512, lane);
      c = wmma_f16(A[kt], b, c);
    }
    store_act(bout, sout, nt, lane, c, bias);
  }
  asm volatile("s_wait_dscnt 0" ::: "memory");
  __builtin_amdgcn_wave_barrier();
}

// Output head: 128 -> nvalid (<=16) via one N-tile of WMMA (weights zero-padded to N=16).
__device__ __forceinline__ void mlp_head(const _Float16* bin, int sin_,
                                         const _Float16* __restrict__ ws, int fragbase,
                                         const float* __restrict__ bias_g, int nvalid,
                                         int out_col_base, float* __restrict__ out,
                                         long long pbase, int lane) {
  const _Float16* row = bin + (lane & 15) * sin_;
  v8f c;
#pragma unroll
  for (int i = 0; i < 8; i++) c[i] = 0.0f;
#pragma unroll
  for (int kt = 0; kt < 4; kt++) {
    v16h a = load_a_frag(row, kt, lane);
    v16h b = load_b_frag(ws, fragbase + kt * 512, lane);
    c = wmma_f16(a, b, c);
  }
  int col = lane & 15;
  if (col < nvalid) {
    float bc = bias_g[col];
    int mhi = (lane >= 16) ? 8 : 0;
#pragma unroll
    for (int v = 0; v < 8; v++)
      out[(pbase + v + mhi) * 4 + out_col_base + col] = c[v] + bc;
  }
}

// ---------------------------------------------------------------------------
// Main fused kernel: 256 threads = 8 waves, 16 points per wave.
// ---------------------------------------------------------------------------
__global__ __launch_bounds__(256)
void fastsurf_kernel(const float* __restrict__ x, const float* __restrict__ grid,
                     const _Float16* __restrict__ wsf,
                     const float* __restrict__ db1, const float* __restrict__ db2,
                     const float* __restrict__ db3,
                     const float* __restrict__ rb1, const float* __restrict__ rb2,
                     const float* __restrict__ rb3,
                     float* __restrict__ out) {
  __shared__ __align__(16) _Float16 sBufA[WAVES][16 * SA];
  __shared__ __align__(16) _Float16 sBufB[WAVES][16 * SB];
  __shared__ float sbias[4][128];

  int tid = threadIdx.x;
  if (tid < 128) {
    sbias[0][tid] = db1[tid];
    sbias[1][tid] = db2[tid];
    sbias[2][tid] = rb1[tid];
    sbias[3][tid] = rb2[tid];
  }
  __syncthreads();

  int lane = tid & 31;
  int wave = tid >> 5;
  int m = lane & 15;
  long long pbase = (long long)blockIdx.x * (WAVES * 16) + wave * 16;
  long long p = pbase + m;

  _Float16* bA = sBufA[wave];
  _Float16* bB = sBufB[wave];
  _Float16* rowA = bA + m * SA;
  const float* xr = x + p * 8;

  if (lane < 16) {
    // ---- trilinear gather: feature[12] -> rgb/sdf input cols 0..11 ----
    float feat[GC];
#pragma unroll
    for (int c = 0; c < GC; c++) feat[c] = 0.0f;
    float pc[3], wf[3];
    int i0[3];
#pragma unroll
    for (int d = 0; d < 3; d++) {
      float u = xr[d];
      u = fminf(fmaxf(u, 0.0f), 1.0f) * (float)(RES - 1);
      int i = (int)floorf(u);
      i = i < 0 ? 0 : (i > RES - 2 ? RES - 2 : i);
      i0[d] = i;
      wf[d] = u - (float)i;
      pc[d] = u;
    }
    (void)pc;
#pragma unroll
    for (int dx = 0; dx < 2; dx++) {
      float wxx = dx ? wf[0] : 1.0f - wf[0];
#pragma unroll
      for (int dy = 0; dy < 2; dy++) {
        float wyy = dy ? wf[1] : 1.0f - wf[1];
        float wz1 = wxx * wyy * wf[2];
        float wz0 = wxx * wyy - wz1;
        const float* cp = grid + (long long)(((i0[0] + dx) * RES + (i0[1] + dy)) * RES + i0[2]) * GC;
        const v4f* q = (const v4f*)cp;  // 48B cells -> 16B aligned
        v4f g0a = q[0], g0b = q[1], g0c = q[2];   // corner z0 (12 floats)
        v4f g1a = q[3], g1b = q[4], g1c = q[5];   // corner z1 (12 floats)
#pragma unroll
        for (int c = 0; c < 4; c++) {
          feat[c]     += wz0 * g0a[c] + wz1 * g1a[c];
          feat[4 + c] += wz0 * g0b[c] + wz1 * g1b[c];
          feat[8 + c] += wz0 * g0c[c] + wz1 * g1c[c];
        }
      }
    }
#pragma unroll
    for (int c = 0; c < GC; c++) rowA[c] = (_Float16)feat[c];
  } else {
    // ---- view embedding + extra features -> cols 12..40, zero-pad to 63 ----
    float vx = xr[5], vy = xr[6], vz = xr[7];
    float f0 = xr[3], f1 = xr[4];
    rowA[12] = (_Float16)vx; rowA[13] = (_Float16)vy; rowA[14] = (_Float16)vz;
    const float FR[4] = {1.0f, 2.0f, 4.0f, 8.0f};
#pragma unroll
    for (int fi = 0; fi < 4; fi++) {
      float fx = vx * FR[fi], fy = vy * FR[fi], fz = vz * FR[fi];
      int c0 = 15 + fi * 6;
      rowA[c0 + 0] = (_Float16)sinf(fx);
      rowA[c0 + 1] = (_Float16)sinf(fy);
      rowA[c0 + 2] = (_Float16)sinf(fz);
      rowA[c0 + 3] = (_Float16)cosf(fx);
      rowA[c0 + 4] = (_Float16)cosf(fy);
      rowA[c0 + 5] = (_Float16)cosf(fz);
    }
    rowA[39] = (_Float16)f0;
    rowA[40] = (_Float16)f1;
#pragma unroll
    for (int c = 41; c < 64; c++) rowA[c] = (_Float16)0.0f;
  }
  asm volatile("s_wait_dscnt 0" ::: "memory");
  __builtin_amdgcn_wave_barrier();

  // ---- SDF path: 12(pad32) -> 128 -> 128 -> 1 ----
  mlp_layer<1>(bA, SA, bB, SB, wsf, FRAG_DW1, sbias[0], lane);
  mlp_layer<4>(bB, SB, bB, SB, wsf, FRAG_DW2, sbias[1], lane);
  mlp_head(bB, SB, wsf, FRAG_DW3, db3, 1, 3, out, pbase, lane);

  // ---- RGB path: 41(pad64) -> 128 -> 128 -> 3 ----
  mlp_layer<2>(bA, SA, bB, SB, wsf, FRAG_RW1, sbias[2], lane);
  mlp_layer<4>(bB, SB, bB, SB, wsf, FRAG_RW2, sbias[3], lane);
  mlp_head(bB, SB, wsf, FRAG_RW3, rb3, 3, 0, out, pbase, lane);
}

extern "C" void kernel_launch(void* const* d_in, const int* in_sizes, int n_in,
                              void* d_out, int out_size, void* d_ws, size_t ws_size,
                              hipStream_t stream) {
  const float* x    = (const float*)d_in[0];
  const float* grid = (const float*)d_in[1];
  const float* dW1  = (const float*)d_in[2];
  const float* db1  = (const float*)d_in[3];
  const float* dW2  = (const float*)d_in[4];
  const float* db2  = (const float*)d_in[5];
  const float* dW3  = (const float*)d_in[6];
  const float* db3  = (const float*)d_in[7];
  const float* rW1  = (const float*)d_in[8];
  const float* rb1  = (const float*)d_in[9];
  const float* rW2  = (const float*)d_in[10];
  const float* rb2  = (const float*)d_in[11];
  const float* rW3  = (const float*)d_in[12];
  const float* rb3  = (const float*)d_in[13];
  _Float16* ws = (_Float16*)d_ws;

  int npts = in_sizes[0] / 8;

  pack_weights<<<(TOTAL_HALVES + 255) / 256, 256, 0, stream>>>(dW1, dW2, dW3, rW1, rW2, rW3, ws);

  int nblocks = npts / (WAVES * 16);
  fastsurf_kernel<<<nblocks, 256, 0, stream>>>(x, grid, ws, db1, db2, db3, rb1, rb2, rb3,
                                               (float*)d_out);
}
